// SignAttention_11072425689793
// MI455X (gfx1250) — matrix operand order repaired and checked by reading, not compiled, not run on hardware
//
#include <hip/hip_runtime.h>
#include <hip/hip_bf16.h>

typedef __bf16 bf16_t;
typedef bf16_t v16bf __attribute__((ext_vector_type(16)));
typedef float  v8f   __attribute__((ext_vector_type(8)));

union FragBF { unsigned int u[8]; v16bf v; };
union FragF  { float        f[8]; v8f   v; };

__device__ __forceinline__ unsigned short f2bf(float f) {
  unsigned int a = __float_as_uint(f);
  return (unsigned short)((a + 0x7FFFu + ((a >> 16) & 1u)) >> 16);
}
__device__ __forceinline__ unsigned int pack_bf16(float lo, float hi) {
  unsigned int a = __float_as_uint(lo);
  unsigned int b = __float_as_uint(hi);
  a = (a + 0x7FFFu + ((a >> 16) & 1u)) >> 16;
  b = (b + 0x7FFFu + ((b >> 16) & 1u)) & 0xFFFF0000u;
  return (a & 0xFFFFu) | b;
}

__device__ __forceinline__ v8f wmma_bf16(v16bf a, v16bf b, v8f c) {
  // D = A(16x32 bf16) * B(32x16 bf16) + C(16x16 f32)
  return __builtin_amdgcn_wmma_f32_16x16x32_bf16(false, a, false, b, (short)0, c,
                                                 false, false);
}

// ---- CDNA5 async global->LDS copy (ASYNCcnt-tracked DMA), with fallback ----
#if __has_builtin(__builtin_amdgcn_global_load_async_to_lds_b128)
#define HAVE_ASYNC_LDS 1
// Builtin signature (from the round-2 diagnostic): param0 is
// 'vector_size(16) int __device__*' (AS1), param1 the AS3 counterpart.
typedef int v4i_bl __attribute__((vector_size(16)));
typedef __attribute__((address_space(1))) v4i_bl* gas_ptr;
typedef __attribute__((address_space(3))) v4i_bl* las_ptr;
__device__ __forceinline__ void async_copy16(const void* g, void* l) {
  __builtin_amdgcn_global_load_async_to_lds_b128(
      (gas_ptr)(unsigned long long)g,
      (las_ptr)(unsigned int)(unsigned long long)l, 0, 0);
}
__device__ __forceinline__ void wait_async() {
#if __has_builtin(__builtin_amdgcn_s_wait_asynccnt)
  __builtin_amdgcn_s_wait_asynccnt(0);
#else
  asm volatile("s_wait_asynccnt 0x0" ::: "memory");
#endif
}
#else
#define HAVE_ASYNC_LDS 0
__device__ __forceinline__ void wait_async() {}
#endif

// A fragment: 16x32 bf16 strip, row-major in LDS (uint row stride ldau).
// Element e of lane L: row = L&15, k = half*16 + e
__device__ __forceinline__ v16bf load_a(const unsigned int* l, int row, int hl, int ldau) {
  FragBF f;
  const unsigned int* p = l + row * ldau;
#pragma unroll
  for (int r = 0; r < 8; ++r) {
    int ku = hl * 8 + r;
    f.u[r] = p[ku];
  }
  return f.v;
}
// B fragment: 32x16 bf16 from n-major ("transposed") LDS tile lds[n][k0..31].
// Element e of lane L: n = L&15, k = half*16 + e.
__device__ __forceinline__ v16bf load_bT(const unsigned int* l, int nrow, int hl, int ldau) {
  FragBF f;
  const unsigned int* p = l + nrow * ldau + hl * 8;
#pragma unroll
  for (int r = 0; r < 8; ++r) f.u[r] = p[r];
  return f.v;
}

// Generic  Out[z] = A(f32 [O,K], shared over z) * B[z] ([K,N])  (+bias)
template <bool B_BF16, bool OUT_F32_BIAS>
__global__ __launch_bounds__(256) void gemm_wmma_kernel(
    const float* __restrict__ A, const void* __restrict__ Bm, void* __restrict__ Out,
    const float* __restrict__ bias, int O, int K, int N, long bStride, long outStride) {
  __shared__ unsigned int lA[128 * 17];  // 128 rows x 32 bf16 (16 dwords + pad)
  __shared__ unsigned int lB[128 * 17];  // 128 n-rows x 32 bf16 (transposed)
  const int tid = threadIdx.x;
  const int wave = tid >> 5, lane = tid & 31;
  const int hl = lane >> 4, l15 = lane & 15;
  const int nb = blockIdx.x * 128, ob = blockIdx.y * 128, z = blockIdx.z;
  const int wo = (wave >> 2) * 64, wn = (wave & 3) * 32;

  const float* Bf = (const float*)Bm + (B_BF16 ? 0 : (size_t)z * bStride);
  const unsigned short* Bh = (const unsigned short*)Bm + (B_BF16 ? (size_t)z * bStride : 0);

  FragF acc[4][2];
#pragma unroll
  for (int i = 0; i < 4; ++i)
#pragma unroll
    for (int j = 0; j < 2; ++j)
#pragma unroll
      for (int r = 0; r < 8; ++r) acc[i][j].f[r] = 0.0f;

  for (int kb = 0; kb < K; kb += 32) {
    __syncthreads();
#pragma unroll
    for (int i = 0; i < 8; ++i) {  // stage A tile [128 x 32] f32 -> bf16
      int p = tid + i * 256;
      int row = p >> 4, ku = p & 15;
      const float* src = A + (size_t)(ob + row) * K + kb + ku * 2;
      lA[row * 17 + ku] = pack_bf16(src[0], src[1]);
      if (i == 0 && kb + 32 < K) __builtin_prefetch(src + 32, 0, 0);
    }
#pragma unroll
    for (int i = 0; i < 8; ++i) {  // stage B tile [32 x 128] -> lB[n][k]
      int p = tid + i * 256;
      int ku = p >> 7, n = p & 127;
      int krow = kb + ku * 2;
      unsigned int v;
      if (B_BF16) {
        unsigned int lo = Bh[(size_t)krow * N + nb + n];
        unsigned int hi = Bh[(size_t)(krow + 1) * N + nb + n];
        v = lo | (hi << 16);
        if (i == 0 && kb + 32 < K)
          __builtin_prefetch(&Bh[(size_t)(krow + 32) * N + nb + n], 0, 0);
      } else {
        v = pack_bf16(Bf[(size_t)krow * N + nb + n], Bf[(size_t)(krow + 1) * N + nb + n]);
        if (i == 0 && kb + 32 < K)
          __builtin_prefetch(&Bf[(size_t)(krow + 32) * N + nb + n], 0, 0);
      }
      lB[n * 17 + ku] = v;
    }
    __syncthreads();
    v16bf afr[4], bfr[2];
#pragma unroll
    for (int mi = 0; mi < 4; ++mi) afr[mi] = load_a(lA, wo + mi * 16 + l15, hl, 17);
#pragma unroll
    for (int ni = 0; ni < 2; ++ni) bfr[ni] = load_bT(lB, wn + ni * 16 + l15, hl, 17);
#pragma unroll
    for (int mi = 0; mi < 4; ++mi)
#pragma unroll
      for (int ni = 0; ni < 2; ++ni)
        acc[mi][ni].v = wmma_bf16(afr[mi], bfr[ni], acc[mi][ni].v);
  }

#pragma unroll
  for (int mi = 0; mi < 4; ++mi)
#pragma unroll
    for (int ni = 0; ni < 2; ++ni)
#pragma unroll
      for (int r = 0; r < 8; ++r) {
        int orow = ob + wo + mi * 16 + r + 8 * hl;
        int ocol = nb + wn + ni * 16 + l15;
        if (OUT_F32_BIAS) {
          float* o = (float*)Out + (size_t)z * outStride;
          o[(size_t)orow * N + ocol] = acc[mi][ni].f[r] + bias[orow];
        } else {
          unsigned short* o = (unsigned short*)Out + (size_t)z * outStride;
          o[(size_t)orow * N + ocol] = f2bf(acc[mi][ni].f[r]);
        }
      }
}

// ---- fused attention ----
// lK rows padded to 20 dwords (80 B): 16B-aligned rows for b128 async DMA and
// conflict-free A-frag reads (20*r mod 64 distinct over 16 consecutive rows).
#define LKS 20
#define LQS 17
#define LPS 65

__device__ __forceinline__ void stage_K(unsigned int* lKbuf, const unsigned short* Kp,
                                        int it, int mb, int tid) {
#if HAVE_ASYNC_LDS
#pragma unroll
  for (int i = 0; i < 2; ++i) {          // 512 x 16B chunks, DMA to LDS
    int c = tid + i * 256;
    int row = c >> 2, seg = c & 3;
    const char* g = (const char*)Kp + (size_t)(it * 128 + row) * 2048 +
                    (size_t)mb * 2 + seg * 16;
    char* l = (char*)lKbuf + row * (LKS * 4) + seg * 16;
    async_copy16(g, l);
  }
#else
  const unsigned int* Kpu = (const unsigned int*)Kp;
#pragma unroll
  for (int i = 0; i < 8; ++i) {
    int p = tid + i * 256;
    int row = p >> 4, ku = p & 15;
    lKbuf[row * LKS + ku] = Kpu[(size_t)(it * 128 + row) * 512 + (mb >> 1) + ku];
  }
#endif
}

__global__ __launch_bounds__(256) void attn_kernel(
    const unsigned short* __restrict__ Kb,   // [B,8192,1024] bf16 row-major
    const unsigned short* __restrict__ Qb,   // [B,8192,1024] bf16 row-major
    const unsigned short* __restrict__ Vb,   // [B,512,1024]  bf16 row-major
    unsigned short* __restrict__ Ub) {       // [B,512,1024]  bf16 row-major
  // P^T aliases the K/Q double buffers (disjoint in time, barrier-separated).
  __shared__ union SharedMem {
    struct { unsigned int k[2][128 * LKS]; unsigned int q[2][128 * LQS]; } kq;
    unsigned int p[128 * LPS];
  } sm;

  const int tid = threadIdx.x;
  const int wave = tid >> 5, lane = tid & 31;
  const int hl = lane >> 4, l15 = lane & 15;
  const int jc = blockIdx.x, h = blockIdx.y, b = blockIdx.z;
  const int jb = jc * 128;

  const size_t kqHead = ((size_t)b * 8192 + (size_t)h * 1024) * 1024;
  const unsigned short* Kp = Kb + kqHead;
  const unsigned short* Qp = Qb + kqHead;

  const int dgrp = (wave >> 1) * 16;  // O-phase d rows
  const int jhw = (wave & 1) * 64;    // O-phase j half

  FragF o[4];
#pragma unroll
  for (int nf = 0; nf < 4; ++nf)
#pragma unroll
    for (int r = 0; r < 8; ++r) o[nf].f[r] = 0.0f;

  for (int it = 0; it < 8; ++it) {  // 128-row i tiles over dk=1024
    FragF s[8];
#pragma unroll
    for (int jf = 0; jf < 8; ++jf)
#pragma unroll
      for (int r = 0; r < 8; ++r) s[jf].f[r] = 0.0f;

    unsigned int qlo[8], qhi[8];

    // prologue: stage tile for mb=0
    stage_K(sm.kq.k[0], Kp, it, 0, tid);
#pragma unroll
    for (int i = 0; i < 8; ++i) {
      int p = tid + i * 256;
      int mu = p >> 7, j = p & 127;
      qlo[i] = Qp[(size_t)(2 * mu) * 1024 + jb + j];
      qhi[i] = Qp[(size_t)(2 * mu + 1) * 1024 + jb + j];
    }
#pragma unroll
    for (int i = 0; i < 8; ++i) {
      int p = tid + i * 256;
      int mu = p >> 7, j = p & 127;
      sm.kq.q[0][j * LQS + mu] = qlo[i] | (qhi[i] << 16);
    }
    wait_async();
    __syncthreads();

    for (int ss = 0; ss < 32; ++ss) {   // 32 m-steps over dk=1024
      const int cur = ss & 1;
      const int mbn = (ss + 1) * 32;
      if (mbn < 1024) {                 // issue next tile: K via async DMA,
        stage_K(sm.kq.k[cur ^ 1], Kp, it, mbn, tid);  // Q loads in flight
#pragma unroll
        for (int i = 0; i < 8; ++i) {
          int p = tid + i * 256;
          int mu = p >> 7, j = p & 127;
          qlo[i] = Qp[(size_t)(mbn + 2 * mu) * 1024 + jb + j];
          qhi[i] = Qp[(size_t)(mbn + 2 * mu + 1) * 1024 + jb + j];
        }
      }
      // compute current tile: 1 A-frag, 8 B-frags, 8 WMMAs (hides the DMA)
      v16bf a = load_a(sm.kq.k[cur], wave * 16 + l15, hl, LKS);
#pragma unroll
      for (int jf = 0; jf < 8; ++jf) {
        v16bf bq = load_bT(sm.kq.q[cur], jf * 16 + l15, hl, LQS);
        s[jf].v = wmma_bf16(a, bq, s[jf].v);
      }
      if (mbn < 1024) {                 // pack + store next Q tile
#pragma unroll
        for (int i = 0; i < 8; ++i) {
          int p = tid + i * 256;
          int mu = p >> 7, j = p & 127;
          sm.kq.q[cur ^ 1][j * LQS + mu] = qlo[i] | (qhi[i] << 16);
        }
      }
      wait_async();
      __syncthreads();
    }

    // Chunk softmax along j (scale 1/sqrt(1024)=1/32 folded into exp).
    // Row m = r + 8*hl lives in the 16 lanes of one half: xor masks 1,2,4,8.
#pragma unroll
    for (int r = 0; r < 8; ++r) {
      float mx = -3.402823466e38f;
#pragma unroll
      for (int jf = 0; jf < 8; ++jf) mx = fmaxf(mx, s[jf].f[r]);
#pragma unroll
      for (int off = 1; off <= 8; off <<= 1) mx = fmaxf(mx, __shfl_xor(mx, off, 32));
      float sum = 0.0f;
#pragma unroll
      for (int jf = 0; jf < 8; ++jf) {
        float pe = __expf((s[jf].f[r] - mx) * 0.03125f);
        s[jf].f[r] = pe;
        sum += pe;
      }
#pragma unroll
      for (int off = 1; off <= 8; off <<= 1) sum += __shfl_xor(sum, off, 32);
      float inv = 1.0f / sum;
#pragma unroll
      for (int jf = 0; jf < 8; ++jf) s[jf].f[r] *= inv;
    }

    {  // store P^T (bf16) to LDS: p[j][i]   (aliases k/q: barrier passed above)
      unsigned short* lPh = (unsigned short*)sm.p;
#pragma unroll
      for (int jf = 0; jf < 8; ++jf) {
        int j = jf * 16 + l15;
#pragma unroll
        for (int r = 0; r < 8; ++r) {
          int irow = wave * 16 + r + 8 * hl;
          lPh[j * (2 * LPS) + irow] = f2bf(s[jf].f[r]);
        }
      }
    }
    __syncthreads();

    // O-phase: o[d, j] += V[d, i-tile] * P[i-tile, j]
    const unsigned int* Vrow =
        (const unsigned int*)(Vb + ((size_t)b * 512 + h * 64) * 1024) +
        (size_t)(dgrp + l15) * 512;
#pragma unroll
    for (int ks = 0; ks < 4; ++ks) {
      FragBF a;
      int kbaseu = (it * 128 + ks * 32) >> 1;
#pragma unroll
      for (int r = 0; r < 8; ++r) {
        int ku = hl * 8 + r;
        a.u[r] = Vrow[kbaseu + ku];
      }
#pragma unroll
      for (int nf = 0; nf < 4; ++nf) {
        FragBF pb;
        const unsigned int* pp =
            sm.p + (size_t)(jhw + nf * 16 + l15) * LPS + ks * 16 + hl * 8;
#pragma unroll
        for (int r = 0; r < 8; ++r) pb.u[r] = pp[r];
        o[nf].v = wmma_bf16(a.v, pb.v, o[nf].v);
      }
    }
    __syncthreads();  // P reads done before next it re-stages the aliased k/q
  }

  unsigned short* Up = Ub + ((size_t)b * 512 + h * 64) * 1024;
#pragma unroll
  for (int nf = 0; nf < 4; ++nf)
#pragma unroll
    for (int r = 0; r < 8; ++r) {
      int d = dgrp + r + 8 * hl;
      int j = jb + jhw + nf * 16 + l15;
      Up[(size_t)d * 1024 + j] = f2bf(o[nf].f[r]);
    }
}

extern "C" void kernel_launch(void* const* d_in, const int* in_sizes, int n_in,
                              void* d_out, int out_size, void* d_ws, size_t ws_size,
                              hipStream_t stream) {
  (void)in_sizes; (void)n_in; (void)out_size; (void)ws_size;
  const float* x  = (const float*)d_in[0];
  const float* Wk = (const float*)d_in[1];
  const float* Wq = (const float*)d_in[2];
  const float* Wv = (const float*)d_in[3];
  const float* Wr = (const float*)d_in[4];
  const float* br = (const float*)d_in[5];

  char* ws = (char*)d_ws;
  const size_t KQ_BYTES = (size_t)8 * 8192 * 1024 * 2;  // 128 MiB
  const size_t V_BYTES  = (size_t)8 * 512 * 1024 * 2;   // 8 MiB
  unsigned short* Kw = (unsigned short*)(ws);
  unsigned short* Qw = (unsigned short*)(ws + KQ_BYTES);
  unsigned short* Vw = (unsigned short*)(ws + 2 * KQ_BYTES);
  unsigned short* Uw = (unsigned short*)(ws + 2 * KQ_BYTES + V_BYTES);

  dim3 blk(256);
  // K = Wk @ X : [8192 x 1024] per batch (bf16 out)
  gemm_wmma_kernel<false, false><<<dim3(8, 64, 8), blk, 0, stream>>>(
      Wk, x, Kw, nullptr, 8192, 256, 1024, (long)256 * 1024, (long)8192 * 1024);
  // Q = Wq @ X
  gemm_wmma_kernel<false, false><<<dim3(8, 64, 8), blk, 0, stream>>>(
      Wq, x, Qw, nullptr, 8192, 256, 1024, (long)256 * 1024, (long)8192 * 1024);
  // V = Wv @ X : [512 x 1024] per batch
  gemm_wmma_kernel<false, false><<<dim3(8, 4, 8), blk, 0, stream>>>(
      Wv, x, Vw, nullptr, 512, 256, 1024, (long)256 * 1024, (long)512 * 1024);
  // Fused scores/softmax/V*P per (j-chunk, head, batch)
  attn_kernel<<<dim3(8, 8, 8), blk, 0, stream>>>(Kw, Qw, Vw, Uw);
  // Y = Wr @ U + br : f32 output
  gemm_wmma_kernel<true, true><<<dim3(8, 2, 8), blk, 0, stream>>>(
      Wr, Uw, d_out, br, 256, 512, 1024, (long)512 * 1024, (long)256 * 1024);
}
